// UpThreeOffsetsConv_87660282511816
// MI455X (gfx1250) — matrix operand order repaired and checked by reading, not compiled, not run on hardware
//
#include <hip/hip_runtime.h>

typedef _Float16 half_t;
typedef __attribute__((ext_vector_type(16))) _Float16     v16h;
typedef __attribute__((ext_vector_type(8)))  float        v8f;
typedef __attribute__((ext_vector_type(4)))  unsigned int u32x4;

union Frag16 { v16h h; u32x4 q[2]; };

#define EPSBN 1e-5f

// ---------------------------------------------------------------------------
// Prep kernel 1: x (NCDHW f32, 64x24^3) -> channels-last f16 xh[z][y][x][c]
// ---------------------------------------------------------------------------
__global__ __launch_bounds__(256) void k_convert_x(const float* __restrict__ x,
                                                   half_t* __restrict__ xh) {
  int idx = blockIdx.x * 256 + threadIdx.x;      // 24*24*24*64 = 884736
  if (idx >= 24 * 24 * 24 * 64) return;
  int c = idx & 63;
  int p = idx >> 6;
  xh[idx] = (half_t)x[c * 13824 + p];
}

// ---------------------------------------------------------------------------
// Prep kernel 2: build fused 192x(27*64) f16 weight slab.
//  rows   0..63  : w1  * inv1                      (plain conv)
//  rows  64..127 : effective deform kernel (a=0.4) * inv2
//  rows 128..191 : effective deform kernel (a=0.7) * inv3
// K ordering: k = tap*64 + cin, tap = az*9+ay*3+ax (offset = a-1)
// ---------------------------------------------------------------------------
__global__ __launch_bounds__(256) void k_build_wcat(
    const float* __restrict__ w1, const float* __restrict__ dw1,
    const float* __restrict__ dw2,
    const float* __restrict__ g1, const float* __restrict__ v1,
    const float* __restrict__ g2, const float* __restrict__ v2,
    const float* __restrict__ g3, const float* __restrict__ v3,
    half_t* __restrict__ Wcat) {
  int idx = blockIdx.x * 256 + threadIdx.x;      // 192*64*27 = 331776
  if (idx >= 192 * 64 * 27) return;
  int t = idx % 27;
  int c = (idx / 27) & 63;
  int j = idx / (27 * 64);
  int az = t / 9, ay = (t % 9) / 3, ax = t % 3;
  float val;
  if (j < 64) {
    int o = j;
    float inv = g1[o] * rsqrtf(v1[o] + EPSBN);
    val = w1[(o * 64 + c) * 27 + t] * inv;
  } else {
    bool br2 = (j < 128);
    int o = j - (br2 ? 64 : 128);
    const float* dw = br2 ? dw1 : dw2;
    float alpha = br2 ? 0.4f : 0.7f;
    // per-dim interp weight F[k][a]: tap k in {0,1,2}, integer offset a-1
    float F[3][3] = {{1.f - alpha, alpha, 0.f},
                     {0.f, 1.f, 0.f},
                     {0.f, alpha, 1.f - alpha}};
    float s = 0.f;
    for (int kz = 0; kz < 3; ++kz) {
      float fz = F[kz][az]; if (fz == 0.f) continue;
      for (int ky = 0; ky < 3; ++ky) {
        float fy = F[ky][ay]; if (fy == 0.f) continue;
        for (int kx = 0; kx < 3; ++kx) {
          float fx = F[kx][ax]; if (fx == 0.f) continue;
          s += dw[(o * 64 + c) * 27 + kz * 9 + ky * 3 + kx] * fz * fy * fx;
        }
      }
    }
    float inv = br2 ? (g2[o] * rsqrtf(v2[o] + EPSBN))
                    : (g3[o] * rsqrtf(v3[o] + EPSBN));
    val = s * inv;
  }
  Wcat[j * 1728 + t * 64 + c] = (half_t)val;
}

// ---------------------------------------------------------------------------
// Prep kernel 3: Wc2 = wc * inv4 (f16), biasc[192], bias2[64]
// ---------------------------------------------------------------------------
__global__ __launch_bounds__(256) void k_build_misc(
    const float* __restrict__ wc,  const float* __restrict__ bc,
    const float* __restrict__ b1,  const float* __restrict__ db1,
    const float* __restrict__ db2,
    const float* __restrict__ g1, const float* __restrict__ be1,
    const float* __restrict__ m1, const float* __restrict__ v1,
    const float* __restrict__ g2, const float* __restrict__ be2,
    const float* __restrict__ m2, const float* __restrict__ v2,
    const float* __restrict__ g3, const float* __restrict__ be3,
    const float* __restrict__ m3, const float* __restrict__ v3,
    const float* __restrict__ g4, const float* __restrict__ be4,
    const float* __restrict__ m4, const float* __restrict__ v4,
    half_t* __restrict__ Wc2, float* __restrict__ biasc,
    float* __restrict__ bias2) {
  int idx = blockIdx.x * 256 + threadIdx.x;      // 64*192 = 12288
  if (idx < 64 * 192) {
    int o = idx / 192;
    float inv4 = g4[o] * rsqrtf(v4[o] + EPSBN);
    Wc2[idx] = (half_t)(wc[idx] * inv4);
  }
  if (idx < 192) {
    float bb;
    if (idx < 64) {
      float inv = g1[idx] * rsqrtf(v1[idx] + EPSBN);
      bb = b1[idx] * inv + be1[idx] - m1[idx] * inv;
    } else if (idx < 128) {
      int o = idx - 64;
      float inv = g2[o] * rsqrtf(v2[o] + EPSBN);
      bb = db1[o] * inv + be2[o] - m2[o] * inv;
    } else {
      int o = idx - 128;
      float inv = g3[o] * rsqrtf(v3[o] + EPSBN);
      bb = db2[o] * inv + be3[o] - m3[o] * inv;
    }
    biasc[idx] = bb;
  }
  if (idx < 64) {
    float inv4 = g4[idx] * rsqrtf(v4[idx] + EPSBN);
    bias2[idx] = bc[idx] * inv4 + be4[idx] - m4[idx] * inv4;
  }
}

// ---------------------------------------------------------------------------
// Main fused kernel: 864 WGs, 256 threads (8 wave32).
// GEMM1: M=128 spatial x N=192 channels, K=27 taps * 64 cin, f16 WMMA.
// Upsample handled by index halving into 24^3 f16 x (L2-resident).
// Epilogue: bias+relu -> cat f16 in LDS -> GEMM2 (A=Wc2[64x192], B=cat)
// -> bias+relu -> coalesced f32 stores (NCDHW).
// ---------------------------------------------------------------------------
__global__ __launch_bounds__(256) void k_fused_main(
    const half_t* __restrict__ xh, const half_t* __restrict__ Wcat,
    const half_t* __restrict__ Wc2, const float* __restrict__ biasc,
    const float* __restrict__ bias2, float* __restrict__ out) {
  __shared__ alignas(16) half_t Asmem[128 * 64];    // 16 KB: per-tap A tile
  __shared__ alignas(16) half_t Csmem[128 * 192];   // 48 KB: cat tile (f16)

  const int tid  = threadIdx.x;
  const int wave = tid >> 5;
  const int lane = tid & 31;
  const int l16  = lane & 15;
  const int m0   = blockIdx.x * 128;                // tile base spatial index

  // staging role: 2 threads per row, 32 channels each
  const int r    = tid >> 1;                        // 0..127
  const int half = tid & 1;
  const int p    = m0 + r;
  const int xg   = p % 48;
  const int yg   = (p / 48) % 48;
  const int zg   = p / 2304;

  // WMMA fragment addressing (per ISA layouts)
  const int mrow = (wave << 4) + l16;               // A row (spatial)
  const int khiA = (lane < 16) ? 0 : 8;             // A lane K-offset pattern
  const int khiB = (lane < 16) ? 0 : 8;             // B lane K-offset pattern

  v8f acc[12];
#pragma unroll
  for (int i = 0; i < 12; ++i)
#pragma unroll
    for (int e = 0; e < 8; ++e) acc[i][e] = 0.f;

  for (int t = 0; t < 27; ++t) {
    const int dz = t / 9 - 1;
    const int dy = (t % 9) / 3 - 1;
    const int dx = t % 3 - 1;

    __syncthreads();
    {   // stage A tile for this tap (zero-pad at borders, upsample via >>1)
      const int zz = zg + dz, yy = yg + dy, xx = xg + dx;
      u32x4 v0 = {0u, 0u, 0u, 0u}, v1 = v0, v2 = v0, v3 = v0;
      if ((unsigned)zz < 48u && (unsigned)yy < 48u && (unsigned)xx < 48u) {
        const u32x4* src = (const u32x4*)(
            xh + ((((zz >> 1) * 24 + (yy >> 1)) * 24 + (xx >> 1)) << 6) +
            (half << 5));
        v0 = src[0]; v1 = src[1]; v2 = src[2]; v3 = src[3];
      }
      u32x4* dst = (u32x4*)(Asmem + (r << 6) + (half << 5));
      dst[0] = v0; dst[1] = v1; dst[2] = v2; dst[3] = v3;
    }
    __syncthreads();

#pragma unroll
    for (int kk = 0; kk < 2; ++kk) {
      Frag16 a;
      const half_t* ap = Asmem + (mrow << 6) + (kk << 5) + khiA;
      a.q[0] = *(const u32x4*)ap;
      a.q[1] = *(const u32x4*)(ap + 16);
      const int kb = t * 64 + kk * 32 + khiB;
#pragma unroll
      for (int nt = 0; nt < 12; ++nt) {
        Frag16 b;
        const half_t* bp = Wcat + (nt * 16 + l16) * 1728 + kb;
        b.q[0] = *(const u32x4*)bp;
        b.q[1] = *(const u32x4*)(bp + 16);
        acc[nt] = __builtin_amdgcn_wmma_f32_16x16x32_f16(
            false, a.h, false, b.h, (short)0, acc[nt], false, false);
      }
    }
  }

  // ---- epilogue 1: bias + relu -> cat tile in LDS (layout [m][192] f16) ----
  const int mbase = (wave << 4) + ((lane < 16) ? 0 : 8);
#pragma unroll
  for (int nt = 0; nt < 12; ++nt) {
    float bj = biasc[nt * 16 + l16];
#pragma unroll
    for (int rr = 0; rr < 8; ++rr) {
      float v = acc[nt][rr] + bj;
      v = v > 0.f ? v : 0.f;
      Csmem[(mbase + rr) * 192 + nt * 16 + l16] = (half_t)v;
    }
  }
  __syncthreads();

  // ---- GEMM2: A = Wc2 (M=64 cout), B = cat (N=128 spatial), K=192 ----
  v8f acc2[4];
#pragma unroll
  for (int i = 0; i < 4; ++i)
#pragma unroll
    for (int e = 0; e < 8; ++e) acc2[i][e] = 0.f;

#pragma unroll
  for (int ks = 0; ks < 6; ++ks) {
    Frag16 b;   // cat column fragment: spatial n = wave*16 + l16
    const half_t* bp = Csmem + ((wave << 4) + l16) * 192 + ks * 32 + khiB;
    b.q[0] = *(const u32x4*)bp;
    b.q[1] = *(const u32x4*)(bp + 16);
#pragma unroll
    for (int mt = 0; mt < 4; ++mt) {
      Frag16 a;  // Wc2 row fragment: cout = mt*16 + l16
      const half_t* ap = Wc2 + (mt * 16 + l16) * 192 + ks * 32 + khiA;
      a.q[0] = *(const u32x4*)ap;
      a.q[1] = *(const u32x4*)(ap + 16);
      acc2[mt] = __builtin_amdgcn_wmma_f32_16x16x32_f16(
          false, a.h, false, b.h, (short)0, acc2[mt], false, false);
    }
  }

  // ---- epilogue 2: bias + relu -> out (NCDHW f32, coalesced per half-wave) --
  const int pcol  = m0 + (wave << 4) + l16;         // spatial index
  const int chalf = (lane < 16) ? 0 : 8;
#pragma unroll
  for (int mt = 0; mt < 4; ++mt) {
#pragma unroll
    for (int rr = 0; rr < 8; ++rr) {
      int cout = mt * 16 + rr + chalf;
      float v = acc2[mt][rr] + bias2[cout];
      v = v > 0.f ? v : 0.f;
      out[cout * 110592 + pcol] = v;
    }
  }
}

// ---------------------------------------------------------------------------
extern "C" void kernel_launch(void* const* d_in, const int* in_sizes, int n_in,
                              void* d_out, int out_size, void* d_ws,
                              size_t ws_size, hipStream_t stream) {
  const float* x   = (const float*)d_in[0];
  const float* w1  = (const float*)d_in[1];
  const float* b1  = (const float*)d_in[2];
  const float* dw1 = (const float*)d_in[3];
  const float* db1 = (const float*)d_in[4];
  const float* dw2 = (const float*)d_in[5];
  const float* db2 = (const float*)d_in[6];
  const float* wc  = (const float*)d_in[7];
  const float* bc  = (const float*)d_in[8];
  const float* g1  = (const float*)d_in[9];
  const float* be1 = (const float*)d_in[10];
  const float* m1  = (const float*)d_in[11];
  const float* v1  = (const float*)d_in[12];
  const float* g2  = (const float*)d_in[13];
  const float* be2 = (const float*)d_in[14];
  const float* m2  = (const float*)d_in[15];
  const float* v2  = (const float*)d_in[16];
  const float* g3  = (const float*)d_in[17];
  const float* be3 = (const float*)d_in[18];
  const float* m3  = (const float*)d_in[19];
  const float* v3  = (const float*)d_in[20];
  const float* g4  = (const float*)d_in[21];
  const float* be4 = (const float*)d_in[22];
  const float* m4  = (const float*)d_in[23];
  const float* v4  = (const float*)d_in[24];

  char* ws = (char*)d_ws;
  half_t* xh    = (half_t*)(ws);                 // 24^3*64*2      = 1,769,472 B
  half_t* Wcat  = (half_t*)(ws + 1769472);       // 192*1728*2    =   663,552 B
  half_t* Wc2   = (half_t*)(ws + 2433024);       // 64*192*2      =    24,576 B
  float*  biasc = (float*)(ws + 2457600);        // 192*4
  float*  bias2 = (float*)(ws + 2458368);        // 64*4
  float*  out   = (float*)d_out;

  k_convert_x<<<dim3(3456), dim3(256), 0, stream>>>(x, xh);
  k_build_wcat<<<dim3(1296), dim3(256), 0, stream>>>(w1, dw1, dw2, g1, v1, g2,
                                                     v2, g3, v3, Wcat);
  k_build_misc<<<dim3(48), dim3(256), 0, stream>>>(
      wc, bc, b1, db1, db2, g1, be1, m1, v1, g2, be2, m2, v2, g3, be3, m3, v3,
      g4, be4, m4, v4, Wc2, biasc, bias2);
  k_fused_main<<<dim3(864), dim3(256), 0, stream>>>(xh, Wcat, Wc2, biasc,
                                                    bias2, out);
}
